// MultiHeadAttention_35699768165157
// MI455X (gfx1250) — compile-verified, matrix operand order checked by reading, not run on hardware
//
#include <hip/hip_runtime.h>
#include <hip/hip_bf16.h>

// Problem constants (from reference): B=2, S=2048, E=512, H=16, HD=32
constexpr int Bc = 2;
constexpr int Sc = 2048;
constexpr int Ec = 512;
constexpr int Hc = 16;
constexpr int HDc = 32;
constexpr int Mc = Bc * Sc;   // 4096 rows for projection GEMMs

typedef __attribute__((ext_vector_type(16))) __bf16 v16bf;
typedef __attribute__((ext_vector_type(8)))  __bf16 v8bf;
typedef __attribute__((ext_vector_type(8)))  float  v8f;
typedef __attribute__((ext_vector_type(4)))  int    v4i;

// Pointer types for the async-to-LDS builtin: (v4i AS1*, v4i AS3*, imm, imm)
typedef __attribute__((address_space(1))) v4i* as1_v4i_p;
typedef __attribute__((address_space(3))) v4i* as3_v4i_p;
typedef const __attribute__((address_space(1))) void* as1_cvoid_p;
typedef __attribute__((address_space(3))) void* as3_void_p;

// Load a 16-element bf16 WMMA A/B fragment for one lane.
// ISA 16-bit A-matrix layout: lane holds row m=L%16; with kb=(L/16)*8 the lane's
// 16 values are k in [kb,kb+8) and [kb+16,kb+24) -> two 16-byte chunks.
__device__ __forceinline__ v16bf load_frag(const __bf16* row_base, int kb) {
  v8bf lo = *(const v8bf*)(row_base + kb);
  v8bf hi = *(const v8bf*)(row_base + kb + 16);
  return __builtin_shufflevector(lo, hi, 0, 1, 2, 3, 4, 5, 6, 7,
                                          8, 9, 10, 11, 12, 13, 14, 15);
}

__device__ __forceinline__ v8f wmma_bf16(v16bf a, v16bf b, v8f c) {
  return __builtin_amdgcn_wmma_f32_16x16x32_bf16(
      /*neg_a=*/false, a, /*neg_b=*/false, b,
      /*c_mod=*/(short)0, c, /*reuse_a=*/false, /*reuse_b=*/false);
}

// ---- gfx1250 async global->LDS copy (16B per lane), ASYNCcnt tracked -------
__device__ __forceinline__ void async_cp16(const __bf16* g, __bf16* l) {
#if __has_builtin(__builtin_amdgcn_global_load_async_to_lds_b128)
  __builtin_amdgcn_global_load_async_to_lds_b128(
      (as1_v4i_p)(as1_cvoid_p)g,   // addrspacecast, then in-AS bitcast (+de-const)
      (as3_v4i_p)(as3_void_p)l, 0, 0);
#else
  unsigned loff = (unsigned)(uintptr_t)l;  // low 32 bits = LDS byte offset
  asm volatile("global_load_async_to_lds_b128 %0, %1, off"
               :: "v"(loff), "v"(g) : "memory");
#endif
}

__device__ __forceinline__ void wait_async_le1() {
#if __has_builtin(__builtin_amdgcn_s_wait_asynccnt)
  __builtin_amdgcn_s_wait_asynccnt(1);
#else
  asm volatile("s_wait_asynccnt 0x1" ::: "memory");
#endif
}
__device__ __forceinline__ void wait_async_le0() {
#if __has_builtin(__builtin_amdgcn_s_wait_asynccnt)
  __builtin_amdgcn_s_wait_asynccnt(0);
#else
  asm volatile("s_wait_asynccnt 0x0" ::: "memory");
#endif
}

// ---------------------------------------------------------------------------
// GEMM: Y[M,N] = A[M,K](fp32) @ W[N,K](fp32)^T   with M=4096, N=K=512
// layout 0: bf16 out, [B,H,S,HD]      (Q, K projections)
// layout 1: bf16 out, [B,H,HD,S]      (V projection, transposed for PV wmma)
// layout 2: fp32 out, row-major [M,N] (final O projection)
// Block: 256 thr (8 waves), tile 128(M) x 64(N), K-step 32.
// Wave w -> rows (w/2)*32, cols (w&1)*32: 2x2 = 4 WMMA accumulators.
// Register-staged pipeline: next K-tile global loads overlap current WMMAs.
// ---------------------------------------------------------------------------
__device__ __forceinline__ void emit_elem(void* outp, int layout, int m, int n,
                                          float val) {
  if (layout == 2) {
    ((float*)outp)[(size_t)m * Ec + n] = val;
  } else {
    int b = m >> 11, s = m & (Sc - 1);
    int h = n >> 5, d = n & (HDc - 1);
    __bf16* o = (__bf16*)outp;
    if (layout == 0)
      o[(((size_t)b * Hc + h) * Sc + s) * HDc + d] = (__bf16)val;
    else
      o[(((size_t)b * Hc + h) * HDc + d) * Sc + s] = (__bf16)val;
  }
}

__global__ __launch_bounds__(256) void gemm_proj_kernel(
    const float* __restrict__ A, const float* __restrict__ W, void* outp,
    int layout) {
  __shared__ __bf16 As[128][40];  // stride 40 elems (80B): bank-spread, 16B aligned
  __shared__ __bf16 Bs[64][40];

  const int tid  = threadIdx.x;
  const int lane = tid & 31;
  const int wave = tid >> 5;
  const int l16  = lane & 15;
  const int hf   = lane >> 4;
  const int kb   = hf * 8;

  const int m0 = blockIdx.x * 128;
  const int n0 = blockIdx.y * 64;
  const int mw = (wave >> 1) * 32;
  const int nw = (wave & 1) * 32;

  v8f c00 = {0.f, 0.f, 0.f, 0.f, 0.f, 0.f, 0.f, 0.f};
  v8f c01 = c00, c10 = c00, c11 = c00;

  float4 ra[4], rb[2];
  // Prologue: load K-tile 0 into registers.
#pragma unroll
  for (int it = 0; it < 4; ++it) {
    int idx = tid + it * 256, r = idx >> 3, c4 = (idx & 7) * 4;
    ra[it] = *(const float4*)&A[(size_t)(m0 + r) * Ec + c4];
  }
#pragma unroll
  for (int it = 0; it < 2; ++it) {
    int idx = tid + it * 256, r = idx >> 3, c4 = (idx & 7) * 4;
    rb[it] = *(const float4*)&W[(size_t)(n0 + r) * Ec + c4];
  }

  for (int k0 = 0; k0 < Ec; k0 += 32) {
    // Commit staged registers to LDS as bf16.
#pragma unroll
    for (int it = 0; it < 4; ++it) {
      int idx = tid + it * 256, r = idx >> 3, c4 = (idx & 7) * 4;
      As[r][c4 + 0] = (__bf16)ra[it].x; As[r][c4 + 1] = (__bf16)ra[it].y;
      As[r][c4 + 2] = (__bf16)ra[it].z; As[r][c4 + 3] = (__bf16)ra[it].w;
    }
#pragma unroll
    for (int it = 0; it < 2; ++it) {
      int idx = tid + it * 256, r = idx >> 3, c4 = (idx & 7) * 4;
      Bs[r][c4 + 0] = (__bf16)rb[it].x; Bs[r][c4 + 1] = (__bf16)rb[it].y;
      Bs[r][c4 + 2] = (__bf16)rb[it].z; Bs[r][c4 + 3] = (__bf16)rb[it].w;
    }
    __syncthreads();

    // Prefetch next K-tile while WMMAs run.
    if (k0 + 32 < Ec) {
#pragma unroll
      for (int it = 0; it < 4; ++it) {
        int idx = tid + it * 256, r = idx >> 3, c4 = (idx & 7) * 4;
        ra[it] = *(const float4*)&A[(size_t)(m0 + r) * Ec + k0 + 32 + c4];
      }
#pragma unroll
      for (int it = 0; it < 2; ++it) {
        int idx = tid + it * 256, r = idx >> 3, c4 = (idx & 7) * 4;
        rb[it] = *(const float4*)&W[(size_t)(n0 + r) * Ec + k0 + 32 + c4];
      }
    }

    v16bf af0 = load_frag(&As[mw + l16][0], kb);
    v16bf af1 = load_frag(&As[mw + 16 + l16][0], kb);
    v16bf bf0 = load_frag(&Bs[nw + l16][0], kb);
    v16bf bf1 = load_frag(&Bs[nw + 16 + l16][0], kb);
    c00 = wmma_bf16(af0, bf0, c00);
    c01 = wmma_bf16(af0, bf1, c01);
    c10 = wmma_bf16(af1, bf0, c10);
    c11 = wmma_bf16(af1, bf1, c11);
    __syncthreads();
  }

  // C-layout: lane half hf holds rows hf*8+j, column l16 within each subtile.
  const int mB0 = m0 + mw + hf * 8;
  const int mB1 = mB0 + 16;
  const int nB0 = n0 + nw + l16;
  const int nB1 = nB0 + 16;
#pragma unroll
  for (int j = 0; j < 8; ++j) {
    emit_elem(outp, layout, mB0 + j, nB0, c00[j]);
    emit_elem(outp, layout, mB0 + j, nB1, c01[j]);
    emit_elem(outp, layout, mB1 + j, nB0, c10[j]);
    emit_elem(outp, layout, mB1 + j, nB1, c11[j]);
  }
}

// ---------------------------------------------------------------------------
// Flash attention, causal + key-padding mask. One wave = 16 queries,
// block = 8 waves = 128 queries. K/V tiles (32 keys) staged cooperatively in
// double-buffered LDS via global_load_async_to_lds_b128: waves 0-3 stage K,
// waves 4-7 stage V (1 async instr per wave per tile -> ASYNCcnt pipeline
// depth 1). All waves run the block's full causal extent (uniform loop, EXEC
// all-ones at every WMMA); fully masked tiles contribute 0.
// Q,K: bf16 [B,H,S,HD]; Vt: bf16 [B,H,HD,S]; ctx out: fp32 [B*S, E].
// ---------------------------------------------------------------------------
__global__ __launch_bounds__(256) void flash_attn_kernel(
    const __bf16* __restrict__ Qb, const __bf16* __restrict__ Kb,
    const __bf16* __restrict__ Vt, const unsigned char* __restrict__ maskp,
    float* __restrict__ ctx) {
  __shared__ __bf16 Ksh[2][32][40];  // [buf][key][hd]   (padded, 16B aligned)
  __shared__ __bf16 Vsh[2][32][40];  // [buf][hd][key]   (V^T tile)
  __shared__ __bf16 Ps[8][16][40];   // per-wave P relay (C-layout -> A-layout)

  const int tid = threadIdx.x;
  const int lane = tid & 31;
  const int wave = tid >> 5;
  const int l16 = lane & 15;
  const int hf = lane >> 4;
  const int kb = hf * 8;

  const int b = blockIdx.z;
  const int h = blockIdx.y;
  const int q0 = blockIdx.x * 128 + wave * 16;

  const __bf16* Qh = Qb + ((size_t)(b * Hc + h)) * Sc * HDc;
  const __bf16* Kh = Kb + ((size_t)(b * Hc + h)) * Sc * HDc;
  const __bf16* Vh = Vt + ((size_t)(b * Hc + h)) * HDc * Sc;
  const unsigned char* mk = maskp + (size_t)b * Sc;

  // Cooperative stage: tid<128 -> K chunk, tid>=128 -> V chunk (16B each).
  const int scid = tid & 127;
  const int srow = scid >> 2;          // 0..31
  const int sc8 = (scid & 3) * 8;      // 0,8,16,24
  const bool stageK = tid < 128;

  // Q fragment (A-matrix 16x32), loaded once.
  const v16bf qf = load_frag(Qh + (size_t)(q0 + l16) * HDc, kb);

  v8f acc0 = {0.f, 0.f, 0.f, 0.f, 0.f, 0.f, 0.f, 0.f};
  v8f acc1 = acc0;
  float mrow[8], lrow[8];
#pragma unroll
  for (int j = 0; j < 8; ++j) { mrow[j] = -3.0e38f; lrow[j] = 0.f; }

  const float SCALE = 0.17677669529663687f;  // 1/sqrt(32)
  const int kt_end = blockIdx.x * 128 + 128;  // block-uniform causal extent

  // Prologue: stage tile 0 into buffer 0.
  if (stageK)
    async_cp16(Kh + (size_t)srow * HDc + sc8, &Ksh[0][srow][sc8]);
  else
    async_cp16(Vh + (size_t)srow * Sc + sc8, &Vsh[0][srow][sc8]);

  int buf = 0;
  for (int kt0 = 0; kt0 < kt_end; kt0 += 32) {
    const bool has_next = (kt0 + 32) < kt_end;
    if (has_next) {
      const int nk = kt0 + 32;
      if (stageK)
        async_cp16(Kh + (size_t)(nk + srow) * HDc + sc8, &Ksh[buf ^ 1][srow][sc8]);
      else
        async_cp16(Vh + (size_t)srow * Sc + nk + sc8, &Vsh[buf ^ 1][srow][sc8]);
      wait_async_le1();   // this wave's current-tile async op has landed
    } else {
      wait_async_le0();
    }
    __syncthreads();      // whole tile (all waves' chunks) visible

    // Scores: S = Q @ K^T (16 queries x 32 keys) via two WMMAs.
    v16bf kf0 = load_frag(&Ksh[buf][l16][0], kb);
    v16bf kf1 = load_frag(&Ksh[buf][16 + l16][0], kb);
    v8f s0 = {0.f, 0.f, 0.f, 0.f, 0.f, 0.f, 0.f, 0.f};
    v8f s1 = s0;
    s0 = wmma_bf16(qf, kf0, s0);
    s1 = wmma_bf16(qf, kf1, s1);

    const int key0 = kt0 + l16;
    const int key1 = kt0 + 16 + l16;
    const bool pad0 = mk[key0] != 0;
    const bool pad1 = mk[key1] != 0;

#pragma unroll
    for (int j = 0; j < 8; ++j) {
      const int q = q0 + hf * 8 + j;
      float v0 = s0[j] * SCALE;
      float v1 = s1[j] * SCALE;
      if (key0 > q || !pad0) v0 = -3.0e38f;
      if (key1 > q || !pad1) v1 = -3.0e38f;

      // Row max over 32 keys: intra-16-lane-half reduction (xor 1,2,4,8).
      float rm = fmaxf(v0, v1);
      rm = fmaxf(rm, __shfl_xor(rm, 1));
      rm = fmaxf(rm, __shfl_xor(rm, 2));
      rm = fmaxf(rm, __shfl_xor(rm, 4));
      rm = fmaxf(rm, __shfl_xor(rm, 8));

      float mnew = fmaxf(mrow[j], rm);
      float p0 = __expf(v0 - mnew);
      float p1 = __expf(v1 - mnew);
      float rs = p0 + p1;
      rs += __shfl_xor(rs, 1);
      rs += __shfl_xor(rs, 2);
      rs += __shfl_xor(rs, 4);
      rs += __shfl_xor(rs, 8);
      float corr = __expf(mrow[j] - mnew);
      lrow[j] = lrow[j] * corr + rs;
      mrow[j] = mnew;
      acc0[j] *= corr;
      acc1[j] *= corr;

      Ps[wave][hf * 8 + j][l16] = (__bf16)p0;
      Ps[wave][hf * 8 + j][16 + l16] = (__bf16)p1;
    }

    // Same-wave LDS store->load ordering for the P relay.
    asm volatile("s_wait_dscnt 0x0" ::: "memory");
    __builtin_amdgcn_wave_barrier();

    v16bf pf = load_frag(&Ps[wave][l16][0], kb);
    v16bf vf0 = load_frag(&Vsh[buf][l16][0], kb);
    v16bf vf1 = load_frag(&Vsh[buf][16 + l16][0], kb);
    acc0 = wmma_bf16(pf, vf0, acc0);
    acc1 = wmma_bf16(pf, vf1, acc1);

    __syncthreads();  // all waves done reading buf before it is re-staged
    buf ^= 1;
  }

  // Normalize and emit context: fp32 [B*S, E], heads interleaved at col h*HD.
#pragma unroll
  for (int j = 0; j < 8; ++j) {
    const int q = q0 + hf * 8 + j;
    const float inv = 1.0f / lrow[j];
    const size_t row = (size_t)(b * Sc + q);
    ctx[row * Ec + h * HDc + l16] = acc0[j] * inv;
    ctx[row * Ec + h * HDc + 16 + l16] = acc1[j] * inv;
  }
}

// ---------------------------------------------------------------------------
extern "C" void kernel_launch(void* const* d_in, const int* in_sizes, int n_in,
                              void* d_out, int out_size, void* d_ws,
                              size_t ws_size, hipStream_t stream) {
  const float* q = (const float*)d_in[0];
  const float* k = (const float*)d_in[1];
  const float* v = (const float*)d_in[2];
  const unsigned char* msk = (const unsigned char*)d_in[3];
  const float* Wq = (const float*)d_in[4];
  const float* Wk = (const float*)d_in[5];
  const float* Wv = (const float*)d_in[6];
  const float* Wo = (const float*)d_in[7];
  float* out = (float*)d_out;

  const size_t nPerProj = (size_t)Bc * Hc * Sc * HDc;  // 2,097,152
  __bf16* Qb = (__bf16*)d_ws;
  __bf16* Kb = Qb + nPerProj;
  __bf16* Vt = Kb + nPerProj;
  float* ctx = (float*)(Vt + nPerProj);

  dim3 gGrid(Mc / 128, Ec / 64, 1);  // (32, 8)
  gemm_proj_kernel<<<gGrid, 256, 0, stream>>>(q, Wq, (void*)Qb, 0);
  gemm_proj_kernel<<<gGrid, 256, 0, stream>>>(k, Wk, (void*)Kb, 0);
  gemm_proj_kernel<<<gGrid, 256, 0, stream>>>(v, Wv, (void*)Vt, 1);

  dim3 aGrid(Sc / 128, Hc, Bc);  // (16, 16, 2)
  flash_attn_kernel<<<aGrid, 256, 0, stream>>>(Qb, Kb, Vt, msk, ctx);

  gemm_proj_kernel<<<gGrid, 256, 0, stream>>>(ctx, Wo, (void*)out, 2);
}
